// SelfAttention_9938554323638
// MI455X (gfx1250) — matrix operand order repaired and checked by reading, not compiled, not run on hardware
//
#include <hip/hip_runtime.h>

// CDNA5 / gfx1250 flash-attention with masked softmax (flash-decoding split).
//  Pass 1: K -> bf16 row-major, V -> bf16 transposed [h][d][seq] in d_ws.
//  Pass 2: one wave32 per (16-row Q tile, K-split); bf16 WMMA for QK^T and PV;
//          online softmax in exp2 domain; emits partial (acc, m, l).
//  Pass 3: merge S partials per row, divide by softmax denominator.

typedef __attribute__((ext_vector_type(16))) __bf16 v16bf;
typedef __attribute__((ext_vector_type(8)))  float  v8f;

#define HEADS 8
#define SEQL  2048
#define HDIM  64
#define NTILES (HEADS * (SEQL / 16))     // 1024 q-tiles
#define WAVES_PER_BLOCK 4

__device__ __forceinline__ float row_max16(float v) {
    // reduce across the 16 lanes of a half-wave (xor 1,2,4,8 stays in half)
    v = fmaxf(v, __shfl_xor(v, 1, 32));
    v = fmaxf(v, __shfl_xor(v, 2, 32));
    v = fmaxf(v, __shfl_xor(v, 4, 32));
    v = fmaxf(v, __shfl_xor(v, 8, 32));
    return v;
}
__device__ __forceinline__ float row_sum16(float v) {
    v += __shfl_xor(v, 1, 32);
    v += __shfl_xor(v, 2, 32);
    v += __shfl_xor(v, 4, 32);
    v += __shfl_xor(v, 8, 32);
    return v;
}

// ---------------------------------------------------------------------------
// Pass 1: K fp32 -> bf16 (row-major), V fp32 -> bf16 transposed [h][d][seq].
// ---------------------------------------------------------------------------
__global__ __launch_bounds__(256)
void prep_kv_kernel(const float* __restrict__ K,
                    const float* __restrict__ V,
                    __bf16* __restrict__ Kbf,
                    __bf16* __restrict__ Vt) {
    __shared__ __bf16 tile[HDIM][HDIM + 8];   // padded to dodge bank conflicts

    const int tid = threadIdx.x;
    const int h   = blockIdx.x / (SEQL / 64);
    const int j0  = (blockIdx.x % (SEQL / 64)) * 64;

    const float* Kh = K + (size_t)h * SEQL * HDIM;
    const float* Vh = V + (size_t)h * SEQL * HDIM;

    #pragma unroll
    for (int e = 0; e < 16; ++e) {
        const int idx = tid + 256 * e;      // 0 .. 4095
        const int j   = idx >> 6;
        const int d   = idx & 63;
        const size_t g = (size_t)(j0 + j) * HDIM + d;
        Kbf[(size_t)h * SEQL * HDIM + g] = (__bf16)Kh[g];
        tile[d][j] = (__bf16)Vh[g];
    }
    __syncthreads();

    #pragma unroll
    for (int e = 0; e < 16; ++e) {
        const int idx = tid + 256 * e;
        const int d   = idx >> 6;
        const int j   = idx & 63;
        Vt[((size_t)h * HDIM + d) * SEQL + j0 + j] = tile[d][j];
    }
}

// ---------------------------------------------------------------------------
// Pass 2: flash attention over one K-split slice.
// ---------------------------------------------------------------------------
__global__ __launch_bounds__(WAVES_PER_BLOCK * 32)
void fa_bf16_wmma_kernel(const float* __restrict__ Q,
                         const __bf16* __restrict__ Kbf,
                         const __bf16* __restrict__ Vt,
                         const int*   __restrict__ Msk,
                         float*       __restrict__ PartAcc,  // [NTILES*S][16][64]
                         float*       __restrict__ PartML,   // [NTILES*S][32] (m,l)
                         float*       __restrict__ O,        // direct-mode output
                         int nsplit, int direct) {
    __shared__ __bf16 pbuf[WAVES_PER_BLOCK][16 * 32];

    const int lane  = threadIdx.x & 31;
    const int wave  = threadIdx.x >> 5;
    const int gw    = blockIdx.x * WAVES_PER_BLOCK + wave;  // (tile, split) id
    const int tile  = gw / nsplit;
    const int split = gw % nsplit;
    const int h     = tile >> 7;                            // 128 q-tiles / head
    const int q0    = (tile & 127) << 4;                    // first q row
    const int span  = SEQL / nsplit;
    const int jbeg  = split * span;
    const int jend  = jbeg + span;

    const float*  Qh = Q   + (size_t)h * SEQL * HDIM;
    const __bf16* Kh = Kbf + (size_t)h * SEQL * HDIM;
    const __bf16* Vh = Vt  + (size_t)h * HDIM * SEQL;       // [d][seq]
    const int*    Mh = Msk + (size_t)h * SEQL * SEQL;

    const int half  = lane >> 4;    // 0: lanes 0-15, 1: lanes 16-31
    const int ln    = lane & 15;
    const int khalf = half * 16;

    // ---- Q as two bf16 A-fragments; fold scale * log2(e) into Q once ----
    const float qscale = 0.125f * 1.44269504088896340736f;  // 1/sqrt(64)*log2e
    v16bf qa[2];
    {
        const int kb = half ? 8 : 0;
        #pragma unroll
        for (int f = 0; f < 2; ++f) {
            const float* qr = Qh + (size_t)(q0 + ln) * HDIM + f * 32;
            #pragma unroll
            for (int r = 0; r < 8; ++r) {
                const int kl = 2 * r + khalf;
                qa[f][2 * r]     = (__bf16)(qr[kl] * qscale);
                qa[f][2 * r + 1] = (__bf16)(qr[kl + 1] * qscale);
            }
        }
    }

    const float NEGBIG = -3.0e38f;   // log2-domain "-inf"; exp2 -> 0
    const v8f   vzero  = {0.f, 0.f, 0.f, 0.f, 0.f, 0.f, 0.f, 0.f};

    float mrow[8], lrow[8], alpha[8];
    v8f acc[4];
    #pragma unroll
    for (int r = 0; r < 8; ++r) { mrow[r] = NEGBIG; lrow[r] = 0.f; }
    #pragma unroll
    for (int nc = 0; nc < 4; ++nc) acc[nc] = vzero;

    __bf16* pl = pbuf[wave];

    for (int j0 = jbeg; j0 < jend; j0 += 32) {
        if (j0 + 32 < jend) {
            __builtin_prefetch(&Mh[(size_t)(q0 + ln) * SEQL + (j0 + 32) + khalf], 0, 0);
        }

        // ---- K B-fragments: pre-converted bf16, 32B contiguous per frag ----
        v16bf kb[2][2];
        #pragma unroll
        for (int t = 0; t < 2; ++t) {
            const __bf16* kr = Kh + (size_t)(j0 + t * 16 + ln) * HDIM + khalf;
            kb[t][0] = *(const v16bf*)(kr);
            kb[t][1] = *(const v16bf*)(kr + 32);
        }

        // ---- S = (Q*qscale) K^T ----
        v8f s0 = vzero, s1 = vzero;
        s0 = __builtin_amdgcn_wmma_f32_16x16x32_bf16(false, qa[0], false, kb[0][0], (short)0, s0, false, false);
        s0 = __builtin_amdgcn_wmma_f32_16x16x32_bf16(false, qa[1], false, kb[0][1], (short)0, s0, false, false);
        s1 = __builtin_amdgcn_wmma_f32_16x16x32_bf16(false, qa[0], false, kb[1][0], (short)0, s1, false, false);
        s1 = __builtin_amdgcn_wmma_f32_16x16x32_bf16(false, qa[1], false, kb[1][1], (short)0, s1, false, false);

        // ---- mask (C/D layout: VGPR r -> row r+8*half, col ln) ----
        #pragma unroll
        for (int r = 0; r < 8; ++r) {
            const int m = r + half * 8;
            const int* mp = Mh + (size_t)(q0 + m) * SEQL + j0 + ln;
            if (mp[0]  == 0) s0[r] = NEGBIG;
            if (mp[16] == 0) s1[r] = NEGBIG;
        }

        // ---- online softmax in exp2 domain ----
        #pragma unroll
        for (int r = 0; r < 8; ++r) {
            const float cm = row_max16(fmaxf(s0[r], s1[r]));
            const float mn = fmaxf(mrow[r], cm);
            const float a  = exp2f(mrow[r] - mn);
            const float p0 = exp2f(s0[r] - mn);
            const float p1 = exp2f(s1[r] - mn);
            lrow[r]  = lrow[r] * a + row_sum16(p0 + p1);
            mrow[r]  = mn;
            alpha[r] = a;
            s0[r] = p0;
            s1[r] = p1;
        }

        #pragma unroll
        for (int nc = 0; nc < 4; ++nc)
            #pragma unroll
            for (int r = 0; r < 8; ++r)
                acc[nc][r] *= alpha[r];

        // ---- P: C/D layout -> LDS -> A layout ----
        #pragma unroll
        for (int r = 0; r < 8; ++r) {
            const int m = r + half * 8;
            pl[m * 32 + ln]      = (__bf16)s0[r];
            pl[m * 32 + 16 + ln] = (__bf16)s1[r];
        }
        asm volatile("s_wait_dscnt 0x0" ::: "memory");

        v16bf pa;
        {
            const int kb2 = half ? 8 : 0;
            #pragma unroll
            for (int r = 0; r < 8; ++r) {
                const int kl = 2 * r + khalf;
                pa[2 * r]     = pl[ln * 32 + kl];
                pa[2 * r + 1] = pl[ln * 32 + kl + 1];
            }
        }
        asm volatile("s_wait_dscnt 0x0" ::: "memory");

        // ---- O += P V : V^T rows give contiguous 32B B-fragments ----
        #pragma unroll
        for (int nc = 0; nc < 4; ++nc) {
            const v16bf vb = *(const v16bf*)(Vh + (size_t)(nc * 16 + ln) * SEQL + j0 + khalf);
            acc[nc] = __builtin_amdgcn_wmma_f32_16x16x32_bf16(false, pa, false, vb, (short)0, acc[nc], false, false);
        }
    }

    if (direct) {
        // single split: finalize in place
        float* Oh = O + (size_t)h * SEQL * HDIM;
        #pragma unroll
        for (int r = 0; r < 8; ++r) {
            const int m = r + half * 8;
            const float inv = 1.0f / lrow[r];
            float* orow = Oh + (size_t)(q0 + m) * HDIM + ln;
            #pragma unroll
            for (int nc = 0; nc < 4; ++nc)
                orow[nc * 16] = acc[nc][r] * inv;
        }
    } else {
        // emit partial (acc, m, l); part id == gw
        #pragma unroll
        for (int r = 0; r < 8; ++r) {
            const int m = r + half * 8;
            float* orow = PartAcc + ((size_t)gw * 16 + m) * HDIM + ln;
            #pragma unroll
            for (int nc = 0; nc < 4; ++nc)
                orow[nc * 16] = acc[nc][r];
        }
        if (ln == 0) {
            #pragma unroll
            for (int r = 0; r < 8; ++r) {
                const int m = r + half * 8;
                PartML[(size_t)gw * 32 + m]      = mrow[r];
                PartML[(size_t)gw * 32 + 16 + m] = lrow[r];
            }
        }
    }
}

// ---------------------------------------------------------------------------
// Pass 3: merge S partials per output row.
// ---------------------------------------------------------------------------
__global__ __launch_bounds__(256)
void fa_merge_kernel(const float* __restrict__ PartAcc,
                     const float* __restrict__ PartML,
                     float* __restrict__ O, int nsplit) {
    const int row  = blockIdx.x * 4 + (threadIdx.x >> 6);   // 0 .. 16383
    const int d    = threadIdx.x & 63;
    const int tile = row >> 4;
    const int rr   = row & 15;

    float mstar = -3.0e38f;
    for (int s = 0; s < nsplit; ++s)
        mstar = fmaxf(mstar, PartML[((size_t)tile * nsplit + s) * 32 + rr]);

    float num = 0.f, den = 0.f;
    for (int s = 0; s < nsplit; ++s) {
        const size_t p = (size_t)tile * nsplit + s;
        const float w = exp2f(PartML[p * 32 + rr] - mstar);
        den += PartML[p * 32 + 16 + rr] * w;
        num += PartAcc[(p * 16 + rr) * HDIM + d] * w;
    }
    O[(size_t)row * HDIM + d] = num / den;
}

extern "C" void kernel_launch(void* const* d_in, const int* in_sizes, int n_in,
                              void* d_out, int out_size, void* d_ws, size_t ws_size,
                              hipStream_t stream) {
    const float* Q   = (const float*)d_in[0];
    const float* K   = (const float*)d_in[1];
    const float* V   = (const float*)d_in[2];
    const int*   Msk = (const int*)d_in[3];
    float*       O   = (float*)d_out;

    // workspace layout: bf16 K | bf16 V^T | PartML | PartAcc
    const size_t kvElems  = (size_t)HEADS * SEQL * HDIM;           // per tensor
    const size_t kvBytes  = 2 * kvElems * sizeof(__bf16);          // 4 MB
    auto needBytes = [&](int S) {
        return kvBytes + (size_t)NTILES * S * 32 * sizeof(float)
                       + (size_t)NTILES * S * 16 * HDIM * sizeof(float);
    };
    // deterministic split choice from fixed ws_size
    int S = 1, direct = 1;
    if (ws_size >= needBytes(4))      { S = 4; direct = 0; }
    else if (ws_size >= needBytes(2)) { S = 2; direct = 0; }
    else if (ws_size >= needBytes(1)) { S = 1; direct = 0; }
    else                              { S = 1; direct = 1; }

    __bf16* Kbf = (__bf16*)d_ws;
    __bf16* Vt  = Kbf + kvElems;
    float* PartML  = (float*)(Vt + kvElems);
    float* PartAcc = PartML + (size_t)NTILES * S * 32;

    hipLaunchKernelGGL(prep_kv_kernel,
                       dim3(HEADS * (SEQL / 64)), dim3(256), 0, stream,
                       K, V, Kbf, Vt);

    const int numWaves = NTILES * S;
    hipLaunchKernelGGL(fa_bf16_wmma_kernel,
                       dim3(numWaves / WAVES_PER_BLOCK),
                       dim3(WAVES_PER_BLOCK * 32), 0, stream,
                       Q, Kbf, Vt, Msk, PartAcc, PartML, O, S, direct);

    if (!direct) {
        const int numRows = NTILES * 16;                            // 16384
        hipLaunchKernelGGL(fa_merge_kernel,
                           dim3(numRows / 4), dim3(256), 0, stream,
                           PartAcc, PartML, O, S);
    }
}
